// TransformerDecoderLayer_68959994904870
// MI455X (gfx1250) — compile-verified, matrix operand order checked by reading, not run on hardware
//
#include <hip/hip_runtime.h>
#include <hip/hip_bf16.h>

// ---------------------------------------------------------------------------
// Problem constants (match reference)
// ---------------------------------------------------------------------------
#define N_TOK 16384
#define B_SCENE 16
#define L_NBR 32
#define D_MODEL 256
#define N_HEAD 8
#define D_HEAD 32
#define D_FF 1024
#define LN_EPS 1e-5f

typedef __attribute__((ext_vector_type(16))) __bf16 v16bf;
typedef __attribute__((ext_vector_type(8)))  __bf16 v8bf;
typedef __attribute__((ext_vector_type(8)))  float  v8f;

// --- Tensor Data Mover availability / arity probing ------------------------
#if defined(__has_builtin)
#  if __has_builtin(__builtin_amdgcn_tensor_load_to_lds)
#    define HAVE_TDM 1
#  endif
#endif
#ifndef HAVE_TDM
#  define HAVE_TDM 0
#endif
#if __has_include(<hip/amd_detail/amd_gfx1250_TDM.h>)
#  define TDM_6ARG 1          // therock-10.0 headers -> clang-23 6-arg builtin
#else
#  define TDM_6ARG 0          // ROCm 7.2 clang-22 -> 5-arg builtin
#endif

typedef __attribute__((ext_vector_type(4))) unsigned u32x4;
typedef __attribute__((ext_vector_type(8))) int      i32x8;
typedef __attribute__((ext_vector_type(4))) int      i32x4;

// f32 -> bf16 round-to-nearest-even
__device__ __forceinline__ __bf16 f2bf(float f) {
    unsigned u = __builtin_bit_cast(unsigned, f);
    unsigned r = u + 0x7FFFu + ((u >> 16) & 1u);
    unsigned short h = (unsigned short)(r >> 16);
    return __builtin_bit_cast(__bf16, h);
}

__device__ __forceinline__ void wait_tensor0() {
#if defined(__has_builtin) && __has_builtin(__builtin_amdgcn_s_wait_tensorcnt)
    __builtin_amdgcn_s_wait_tensorcnt(0);
#else
    asm volatile("s_wait_tensorcnt 0x0" ::: "memory");
#endif
}

#if HAVE_TDM
// ---------------------------------------------------------------------------
// TDM: DMA a 2-D bf16 tile [tile_rows x tile_k] (row stride = stride_elems)
// from global memory into LDS at byte offset lds_off.  D# per CDNA5 ISA §8.
// ---------------------------------------------------------------------------
__device__ __forceinline__ void tdm_load_tile(const void* gaddr, unsigned lds_off,
                                              int stride_elems, int tile_k,
                                              int tile_rows) {
    unsigned long long ga = (unsigned long long)gaddr;
    u32x4 g0;
    g0[0] = 1u;                                            // count=1 (valid), user D#
    g0[1] = lds_off;                                       // lds_addr (bytes)
    g0[2] = (unsigned)ga;                                  // global_addr[31:0]
    g0[3] = (unsigned)((ga >> 32) & 0x01FFFFFFull)         // global_addr[56:32]
          | (2u << 30);                                    // type = 2 ("image")
    i32x8 g1;
    g1[0] = 0x00010000;                                    // wg_mask=0, data_size=2B
    g1[1] = (int)((unsigned)(tile_k & 0xFFFF) << 16);      // tensor_dim0[15:0]
    g1[2] = (int)(((unsigned)tile_k >> 16)                 // tensor_dim0[31:16]
          | ((unsigned)(tile_rows & 0xFFFF) << 16));       // tensor_dim1[15:0]
    g1[3] = (int)((unsigned)(tile_k & 0xFFFF) << 16);      // dim1 hi=0 | tile_dim0
    g1[4] = tile_rows & 0xFFFF;                            // tile_dim1 | tile_dim2=0
    g1[5] = stride_elems;                                  // tensor_dim0_stride[31:0]
    g1[6] = 0;                                             // stride hi | dim1_stride
    g1[7] = 0;
    i32x4 z4 = {0, 0, 0, 0};
#if TDM_6ARG
    i32x8 z8 = {0, 0, 0, 0, 0, 0, 0, 0};
    __builtin_amdgcn_tensor_load_to_lds(g0, g1, z4, z4, z8, 0);
#else
    __builtin_amdgcn_tensor_load_to_lds(g0, g1, z4, z4, 0);
#endif
}
#endif

// ---------------------------------------------------------------------------
// Elementwise f32 -> bf16 convert (inputs/weights one-shot pass)
// ---------------------------------------------------------------------------
__global__ __launch_bounds__(256)
void cvt_bf16_kernel(const float* __restrict__ in, __bf16* __restrict__ out,
                     long long n8) {
    long long i = (long long)blockIdx.x * 256 + threadIdx.x;
    if (i >= n8) return;
    const float4* p = (const float4*)in + i * 2;
    float4 a = p[0], b = p[1];
    v8bf o;
    o[0] = f2bf(a.x); o[1] = f2bf(a.y); o[2] = f2bf(a.z); o[3] = f2bf(a.w);
    o[4] = f2bf(b.x); o[5] = f2bf(b.y); o[6] = f2bf(b.z); o[7] = f2bf(b.w);
    ((v8bf*)out)[i] = o;
}

// ---------------------------------------------------------------------------
// WMMA bf16 fragment loader (wave32 layout, CDNA5 ISA 7.12.2).
//   lanes 0-15 : row = row0+lane,    elems 0-7 -> K k0+0..7,  8-15 -> K k0+16..23
//   lanes 16-31: row = row0+lane-16, elems 0-7 -> K k0+8..15, 8-15 -> K k0+24..31
// Works for global or LDS sources (address space inferred after inlining).
// ---------------------------------------------------------------------------
__device__ __forceinline__ v16bf load_frag(const __bf16* __restrict__ X,
                                           int ld, int row0, int k0, int lane) {
    const int half = lane >> 4;
    const int r    = row0 + (lane & 15);
    const __bf16* p = X + (size_t)r * ld + k0 + half * 8;
    v8bf lo = *(const v8bf*)p;          // K block +0
    v8bf hi = *(const v8bf*)(p + 16);   // K block +16
    return __builtin_shufflevector(lo, hi, 0, 1, 2, 3, 4, 5, 6, 7,
                                   8, 9, 10, 11, 12, 13, 14, 15);
}

// ---------------------------------------------------------------------------
// GEMM:  out[M x Dout] = A[M x DIN] @ W[Dout x DIN]^T + bias  (bf16 in, f32 acc)
// DUAL adds A2 @ W2^T (+ bias2) into the same accumulators.
// Block = 8 waves (32x8): block tile 128 x 64; each wave owns 16 x 64.
// W block-column staged in LDS in K-chunks of 256 via the Tensor Data Mover.
// B fragments are preloaded as a batch (8x ds_load_b128 clause) so the four
// WMMAs issue back-to-back behind a single dscnt wait.
// grid = (M/128, Dout/64)
// ---------------------------------------------------------------------------
template <int DIN, bool DUAL, bool RELU, bool OUT_BF16>
__global__ __launch_bounds__(256)
void gemm_wmma(const __bf16* __restrict__ A,  const __bf16* __restrict__ A2,
               const __bf16* __restrict__ W,  const __bf16* __restrict__ W2,
               const float* __restrict__ bias, const float* __restrict__ bias2,
               float* __restrict__ outf, __bf16* __restrict__ outh, int Dout) {
    constexpr int KC = 256;                       // K-chunk staged in LDS
    __shared__ __bf16 ldsW[(DUAL ? 2 : 1) * 64 * KC];

    const int lane = threadIdx.x;                 // 0..31
    const int wave = threadIdx.y;                 // 0..7
    const int row0 = blockIdx.x * 128 + wave * 16;
    const int colb = blockIdx.y * 64;

    v8f c[4];
#pragma unroll
    for (int t = 0; t < 4; ++t) c[t] = (v8f){0.f,0.f,0.f,0.f,0.f,0.f,0.f,0.f};

    for (int kc = 0; kc < DIN; kc += KC) {
        __syncthreads();   // previous chunk fully consumed before overwrite
#if HAVE_TDM
        if (wave == 0) {
            const unsigned wbase = (unsigned)(unsigned long long)(size_t)&ldsW[0];
            tdm_load_tile(W + (size_t)colb * DIN + kc, wbase, DIN, KC, 64);
            if (DUAL)
                tdm_load_tile(W2 + (size_t)colb * DIN + kc, wbase + 64 * KC * 2,
                              DIN, KC, 64);
            wait_tensor0();
        }
#else
        {   // cooperative fallback copy: 64 rows x KC bf16 per operand
            const int tid = wave * 32 + lane;
            for (int i = tid; i < 64 * (KC / 8); i += 256) {
                int r = i >> 5, j = i & 31;
                ((v8bf*)ldsW)[(size_t)r * (KC / 8) + j] =
                    *(const v8bf*)(W + (size_t)(colb + r) * DIN + kc + j * 8);
            }
            if (DUAL) {
                for (int i = tid; i < 64 * (KC / 8); i += 256) {
                    int r = i >> 5, j = i & 31;
                    ((v8bf*)ldsW)[64 * (KC / 8) + (size_t)r * (KC / 8) + j] =
                        *(const v8bf*)(W2 + (size_t)(colb + r) * DIN + kc + j * 8);
                }
            }
        }
#endif
        __syncthreads();

#pragma unroll 4
        for (int k0 = 0; k0 < KC; k0 += 32) {
            if (k0 + 32 < KC)   // prefetch next K tile of the activation rows
                __builtin_prefetch(A + (size_t)(row0 + (lane & 15)) * DIN
                                     + kc + k0 + 32, 0, 1);

            // batch-load all operands first, then issue WMMAs back-to-back
            v16bf a = load_frag(A, DIN, row0, kc + k0, lane);
            v16bf b[4];
#pragma unroll
            for (int t = 0; t < 4; ++t)
                b[t] = load_frag(ldsW, KC, 16 * t, k0, lane);
#pragma unroll
            for (int t = 0; t < 4; ++t)
                c[t] = __builtin_amdgcn_wmma_f32_16x16x32_bf16(
                           false, a, false, b[t], (short)0, c[t], false, false);

            if (DUAL) {
                v16bf a2 = load_frag(A2, DIN, row0, kc + k0, lane);
                v16bf b2[4];
#pragma unroll
                for (int t = 0; t < 4; ++t)
                    b2[t] = load_frag(ldsW + 64 * KC, KC, 16 * t, k0, lane);
#pragma unroll
                for (int t = 0; t < 4; ++t)
                    c[t] = __builtin_amdgcn_wmma_f32_16x16x32_bf16(
                               false, a2, false, b2[t], (short)0, c[t], false, false);
            }
        }
    }

    // C/D layout: lanes 0-15 -> N=lane, M=elem; lanes 16-31 -> N=lane-16, M=elem+8
    const int n     = lane & 15;
    const int mbase = (lane >> 4) * 8;
#pragma unroll
    for (int t = 0; t < 4; ++t) {
        const int col0 = colb + 16 * t;
        float bsum = bias[col0 + n] + (DUAL ? bias2[col0 + n] : 0.f);
#pragma unroll
        for (int r = 0; r < 8; ++r) {
            float x = c[t][r] + bsum;
            if (RELU) x = fmaxf(x, 0.f);
            const size_t idx = (size_t)(row0 + mbase + r) * Dout + col0 + n;
            if (OUT_BF16) outh[idx] = f2bf(x);
            else          outf[idx] = x;
        }
    }
}

// ---------------------------------------------------------------------------
// Gathered local attention.  One wave32 per query; lane = neighbor index l.
// Output stored directly in bf16 (consumed only by the out-proj WMMA GEMM).
// ---------------------------------------------------------------------------
__global__ __launch_bounds__(256)
void attn_kernel(const float* __restrict__ q, const float* __restrict__ k,
                 const float* __restrict__ v,
                 const int* __restrict__ index_pair,
                 const int* __restrict__ index_batch,
                 const int* __restrict__ key_cnt,
                 __bf16* __restrict__ out) {
    __shared__ float qs[8][D_MODEL];

    const int lane = threadIdx.x;               // 0..31 == neighbor l
    const int wave = threadIdx.y;               // 0..7
    const int nq   = blockIdx.x * 8 + wave;

#pragma unroll
    for (int j = 0; j < 8; ++j)
        qs[wave][lane + j * 32] = q[(size_t)nq * D_MODEL + lane + j * 32];
    __syncthreads();

    const int b = index_batch[nq];
    int off = 0;
    for (int i = 0; i < b; ++i) off += key_cnt[i];

    const int  ip     = index_pair[(size_t)nq * L_NBR + lane];
    const bool masked = (ip < 0);
    const int  gidx   = masked ? 0 : (ip + off);

    const float scale = 0.17677669529663687f;   // 32^-0.5

    float s[N_HEAD];
    const float* krow = k + (size_t)gidx * D_MODEL;
#pragma unroll
    for (int h = 0; h < N_HEAD; ++h) {
        float acc = 0.f;
#pragma unroll
        for (int d = 0; d < D_HEAD; ++d)
            acc = fmaf(qs[wave][h * D_HEAD + d], krow[h * D_HEAD + d], acc);
        s[h] = masked ? -1e9f : acc * scale;
    }

    float p[N_HEAD];
#pragma unroll
    for (int h = 0; h < N_HEAD; ++h) {
        float m = s[h];
#pragma unroll
        for (int o = 16; o > 0; o >>= 1) m = fmaxf(m, __shfl_xor(m, o, 32));
        float e = __expf(s[h] - m);
        float sum = e;
#pragma unroll
        for (int o = 16; o > 0; o >>= 1) sum += __shfl_xor(sum, o, 32);
        p[h] = e / sum;
    }

    float acc[N_HEAD];
#pragma unroll
    for (int h = 0; h < N_HEAD; ++h) acc[h] = 0.f;

    for (int l = 0; l < L_NBR; ++l) {
        const int gl = __shfl(gidx, l, 32);
        const float* vrow = v + (size_t)gl * D_MODEL;
#pragma unroll
        for (int h = 0; h < N_HEAD; ++h) {
            const float pv = __shfl(p[h], l, 32);
            acc[h] = fmaf(pv, vrow[h * D_HEAD + lane], acc[h]);
        }
    }
#pragma unroll
    for (int h = 0; h < N_HEAD; ++h)
        out[(size_t)nq * D_MODEL + h * D_HEAD + lane] = f2bf(acc[h]);
}

// ---------------------------------------------------------------------------
// out = LayerNorm(a + b) * g + beta   (row width 256, one wave per row).
// Optionally also emits a bf16 copy in the same pass (operand for next GEMM).
// ---------------------------------------------------------------------------
__global__ __launch_bounds__(256)
void add_ln_kernel(const float* __restrict__ a, const float* __restrict__ bsrc,
                   const float* __restrict__ g, const float* __restrict__ be,
                   float* __restrict__ out, __bf16* __restrict__ out_h) {
    const int lane = threadIdx.x;
    const int wave = threadIdx.y;
    const int row  = blockIdx.x * 8 + wave;
    const size_t base = (size_t)row * D_MODEL;

    float x[8];
    float sum = 0.f;
#pragma unroll
    for (int j = 0; j < 8; ++j) {
        int i = lane + j * 32;
        x[j] = a[base + i] + bsrc[base + i];
        sum += x[j];
    }
#pragma unroll
    for (int o = 16; o > 0; o >>= 1) sum += __shfl_xor(sum, o, 32);
    const float mean = sum * (1.0f / D_MODEL);

    float vs = 0.f;
#pragma unroll
    for (int j = 0; j < 8; ++j) {
        float d = x[j] - mean;
        vs = fmaf(d, d, vs);
    }
#pragma unroll
    for (int o = 16; o > 0; o >>= 1) vs += __shfl_xor(vs, o, 32);
    const float inv = rsqrtf(vs * (1.0f / D_MODEL) + LN_EPS);

#pragma unroll
    for (int j = 0; j < 8; ++j) {
        int i = lane + j * 32;
        float y = (x[j] - mean) * inv * g[i] + be[i];
        out[base + i] = y;
        if (out_h) out_h[base + i] = f2bf(y);
    }
}

// ---------------------------------------------------------------------------
// Host-side launcher
// ---------------------------------------------------------------------------
static inline void launch_cvt(const float* in, __bf16* out, long long n,
                              hipStream_t stream) {
    long long n8 = n / 8;
    int grid = (int)((n8 + 255) / 256);
    cvt_bf16_kernel<<<grid, 256, 0, stream>>>(in, out, n8);
}

extern "C" void kernel_launch(void* const* d_in, const int* in_sizes, int n_in,
                              void* d_out, int out_size, void* d_ws, size_t ws_size,
                              hipStream_t stream) {
    const float* tgt  = (const float*)d_in[0];
    const float* mem  = (const float*)d_in[1];
    const float* pos  = (const float*)d_in[2];
    const int*   kcnt = (const int*)d_in[3];
    const int*   ipair  = (const int*)d_in[4];
    const int*   ibatch = (const int*)d_in[5];
    const float* Wq  = (const float*)d_in[6];  const float* bq  = (const float*)d_in[7];
    const float* Wkc = (const float*)d_in[8];  const float* bkc = (const float*)d_in[9];
    const float* Wkp = (const float*)d_in[10]; const float* bkp = (const float*)d_in[11];
    const float* Wv  = (const float*)d_in[12]; const float* bv  = (const float*)d_in[13];
    const float* Wo  = (const float*)d_in[14]; const float* bo  = (const float*)d_in[15];
    const float* W1  = (const float*)d_in[16]; const float* b1  = (const float*)d_in[17];
    const float* W2  = (const float*)d_in[18]; const float* b2  = (const float*)d_in[19];
    const float* g2  = (const float*)d_in[20]; const float* be2 = (const float*)d_in[21];
    const float* g3  = (const float*)d_in[22]; const float* be3 = (const float*)d_in[23];
    (void)in_sizes; (void)n_in; (void)out_size; (void)ws_size;

    char* wsb = (char*)d_ws;
    const long long ND = (long long)N_TOK * D_MODEL;   // 4,194,304
    const long long WD = (long long)D_MODEL * D_MODEL; //    65,536
    const long long WF = (long long)D_FF * D_MODEL;    //   262,144

    // ---- workspace layout (MB offsets; regions reused once dead) ----------
    __bf16* tgt_h = (__bf16*)(wsb + (0ll  << 20));   //  8 MB (dies after q proj)
    __bf16* mem_h = (__bf16*)(wsb + (8ll  << 20));   //  8 MB (dies after v proj)
    __bf16* pos_h = (__bf16*)(wsb + (16ll << 20));   //  8 MB (dies after k proj)
    __bf16* Wq_h  = (__bf16*)(wsb + (24ll << 20));
    __bf16* Wkc_h = Wq_h  + WD;
    __bf16* Wkp_h = Wkc_h + WD;
    __bf16* Wv_h  = Wkp_h + WD;
    __bf16* Wo_h  = Wv_h  + WD;
    __bf16* W1_h  = Wo_h  + WD;
    __bf16* W2_h  = W1_h  + WF;                      // weights end < 26 MB
    float*  qb    = (float*)(wsb + (26ll << 20));    // 16 MB
    float*  kb    = (float*)(wsb + (42ll << 20));    // 16 MB
    float*  vb    = (float*)(wsb + (58ll << 20));    // 16 MB
    __bf16* ab_h  = (__bf16*)(wsb + (74ll << 20));   //  8 MB attention out (bf16)
    float*  t2    = (float*)(wsb + (26ll << 20));    // reuse qb
    float*  xb    = (float*)(wsb + (42ll << 20));    // reuse kb (live to end)
    __bf16* x_h   = (__bf16*)(wsb + (8ll  << 20));   // reuse mem_h
    __bf16* h1    = (__bf16*)(wsb + (58ll << 20));   // 32 MB, reuse vb..ab_h (peak 90 MB)
    float*  ff    = (float*)(wsb + (26ll << 20));    // reuse t2

    dim3 blk(32, 8);
    dim3 g_d(N_TOK / 128, D_MODEL / 64);   // (128, 4)
    dim3 g_ff(N_TOK / 128, D_FF / 64);     // (128, 16)
    dim3 rows(N_TOK / 8);                  // 2048

    // 0) one-shot bf16 conversions (weights + input activations)
    launch_cvt(tgt, tgt_h, ND, stream);
    launch_cvt(mem, mem_h, ND, stream);
    launch_cvt(pos, pos_h, ND, stream);
    launch_cvt(Wq,  Wq_h,  WD, stream);
    launch_cvt(Wkc, Wkc_h, WD, stream);
    launch_cvt(Wkp, Wkp_h, WD, stream);
    launch_cvt(Wv,  Wv_h,  WD, stream);
    launch_cvt(Wo,  Wo_h,  WD, stream);
    launch_cvt(W1,  W1_h,  WF, stream);
    launch_cvt(W2,  W2_h,  WF, stream);

    // 1) projections
    gemm_wmma<D_MODEL, false, false, false><<<g_d, blk, 0, stream>>>(
        tgt_h, nullptr, Wq_h, nullptr, bq, nullptr, qb, nullptr, D_MODEL);
    gemm_wmma<D_MODEL, true,  false, false><<<g_d, blk, 0, stream>>>(
        mem_h, pos_h, Wkc_h, Wkp_h, bkc, bkp, kb, nullptr, D_MODEL);
    gemm_wmma<D_MODEL, false, false, false><<<g_d, blk, 0, stream>>>(
        mem_h, nullptr, Wv_h, nullptr, bv, nullptr, vb, nullptr, D_MODEL);

    // 2) gathered local attention (bf16 output, ready for out-proj GEMM)
    attn_kernel<<<rows, blk, 0, stream>>>(qb, kb, vb, ipair, ibatch, kcnt, ab_h);

    // 3) out projection
    gemm_wmma<D_MODEL, false, false, false><<<g_d, blk, 0, stream>>>(
        ab_h, nullptr, Wo_h, nullptr, bo, nullptr, t2, nullptr, D_MODEL);

    // 4) x = LN(tgt + tgt2)  (f32 residual copy + bf16 GEMM operand, one pass)
    add_ln_kernel<<<rows, blk, 0, stream>>>(tgt, t2, g2, be2, xb, x_h);

    // 5) FFN up + ReLU (store bf16 directly for FFN down)
    gemm_wmma<D_MODEL, false, true, true><<<g_ff, blk, 0, stream>>>(
        x_h, nullptr, W1_h, nullptr, b1, nullptr, nullptr, h1, D_FF);

    // 6) FFN down
    gemm_wmma<D_FF, false, false, false><<<g_d, blk, 0, stream>>>(
        h1, nullptr, W2_h, nullptr, b2, nullptr, ff, nullptr, D_MODEL);

    // 7) out = LN(x + ff)
    add_ln_kernel<<<rows, blk, 0, stream>>>(xb, ff, g3, be3, (float*)d_out, nullptr);
}
